// BaselineAEAttention_88038239634070
// MI455X (gfx1250) — compile-verified, matrix operand order checked by reading, not run on hardware
//
#include <hip/hip_runtime.h>
#include <hip/hip_bf16.h>
#include <math.h>

// ---------------------------------------------------------------------------
// Problem constants (from the reference): B=32, N=1024, F=256, FFN=1024,
// D=512, H=256.  M = B*N = 32768 token rows.
// ---------------------------------------------------------------------------
#define MTOK   32768
#define FDIM   256
#define FFNDIM 1024
#define DDIM   512
#define HDIM   256
#define NSEQ   1024
#define BATCH  32
#define LNEPS  1e-5f

typedef __bf16 bf16_t;
typedef __attribute__((ext_vector_type(16))) __bf16 bf16x16;
typedef __attribute__((ext_vector_type(8)))  float  f32x8;
typedef unsigned int u32;
typedef __attribute__((ext_vector_type(4)))  u32    u32x4;

union Frag { bf16x16 v; u32x4 q[2]; };

// ---------------------------------------------------------------------------
// Generic bf16 WMMA GEMM:  out[M,Nout] = epilogue(A[M,K] @ W[Nout,K]^T + bias)
// Wave32: each wave computes a 32x32 tile = 2x2 v_wmma_f32_16x16x32_bf16
// fragments with A/B register reuse.  K-contiguous rows on both operands ->
// each lane's fragment is two 16B vector loads per matrix per K-step.
// EPI: 0 = linear -> f32, 1 = exact GELU -> f32, 2 = ReLU -> bf16
// ---------------------------------------------------------------------------
template<int EPI>
__device__ __forceinline__ void store_tile(f32x8 acc, int m0, int n0,
                                           const float* __restrict__ bias,
                                           float* __restrict__ outF,
                                           bf16_t* __restrict__ outB,
                                           int ldout, int lane) {
  const int n  = n0 + (lane & 15);
  const int mb = m0 + ((lane >> 4) << 3);   // lanes 16-31 hold rows r+8
  const float bv = bias ? bias[n] : 0.0f;
#pragma unroll
  for (int r = 0; r < 8; ++r) {
    float v = acc[r] + bv;
    size_t idx = (size_t)(mb + r) * ldout + n;
    if (EPI == 1) {
      v = 0.5f * v * (1.0f + erff(v * 0.70710678118654752f));  // exact GELU
      outF[idx] = v;
    } else if (EPI == 2) {
      outB[idx] = (bf16_t)fmaxf(v, 0.0f);
    } else {
      outF[idx] = v;
    }
  }
}

template<int EPI>
__global__ __launch_bounds__(256) void gemm_wmma_bf16(
    const bf16_t* __restrict__ A,   // [M, K] row-major
    const bf16_t* __restrict__ W,   // [Nout, K] row-major (B = W^T)
    const float*  __restrict__ bias,// [Nout]
    float*  __restrict__ outF,
    bf16_t* __restrict__ outB,
    int M, int K, int Nout, int ldout)
{
  const int lane = threadIdx.x & 31;
  const int wave = threadIdx.x >> 5;
  const int nt   = Nout >> 5;                       // 32-wide N tiles
  const int tile = blockIdx.x * 8 + wave;
  const int tiles = (M >> 5) * nt;
  if (tile >= tiles) return;
  const int tm = (tile / nt) << 5;
  const int tn = (tile % nt) << 5;

  const int half = lane >> 4;
  const int lrow = lane & 15;

  const bf16_t* a0 = A + (size_t)(tm + lrow)      * K;
  const bf16_t* a1 = A + (size_t)(tm + 16 + lrow) * K;
  const bf16_t* b0 = W + (size_t)(tn + lrow)      * K;
  const bf16_t* b1 = W + (size_t)(tn + 16 + lrow) * K;

  f32x8 acc00 = {}, acc01 = {}, acc10 = {}, acc11 = {};

  for (int k0 = 0; k0 < K; k0 += 32) {
    // 16-bit A layout: lanes 0-15 hold K {0..7,16..23}, lanes 16-31 {8..15,24..31}
    const int kb = k0 + half * 8;
    Frag fa0, fa1, fb0, fb1;
    fa0.q[0] = *(const u32x4*)(a0 + kb); fa0.q[1] = *(const u32x4*)(a0 + kb + 16);
    fa1.q[0] = *(const u32x4*)(a1 + kb); fa1.q[1] = *(const u32x4*)(a1 + kb + 16);
    fb0.q[0] = *(const u32x4*)(b0 + kb); fb0.q[1] = *(const u32x4*)(b0 + kb + 16);
    fb1.q[0] = *(const u32x4*)(b1 + kb); fb1.q[1] = *(const u32x4*)(b1 + kb + 16);

    acc00 = __builtin_amdgcn_wmma_f32_16x16x32_bf16(false, fa0.v, false, fb0.v, (short)0, acc00, false, false);
    acc01 = __builtin_amdgcn_wmma_f32_16x16x32_bf16(false, fa0.v, false, fb1.v, (short)0, acc01, false, false);
    acc10 = __builtin_amdgcn_wmma_f32_16x16x32_bf16(false, fa1.v, false, fb0.v, (short)0, acc10, false, false);
    acc11 = __builtin_amdgcn_wmma_f32_16x16x32_bf16(false, fa1.v, false, fb1.v, (short)0, acc11, false, false);
  }

  store_tile<EPI>(acc00, tm,      tn,      bias, outF, outB, ldout, lane);
  store_tile<EPI>(acc01, tm,      tn + 16, bias, outF, outB, ldout, lane);
  store_tile<EPI>(acc10, tm + 16, tn,      bias, outF, outB, ldout, lane);
  store_tile<EPI>(acc11, tm + 16, tn + 16, bias, outF, outB, ldout, lane);
}

// ---------------------------------------------------------------------------
// Helpers: f32 -> bf16 conversion, block reduction
// ---------------------------------------------------------------------------
__global__ void cvt_f32_to_bf16(const float* __restrict__ in,
                                bf16_t* __restrict__ out, long n) {
  long i = (long)blockIdx.x * blockDim.x + threadIdx.x;
  if (i < n) out[i] = (bf16_t)in[i];
}

__device__ __forceinline__ float block_reduce(float v, float* sh) {
  int tid = threadIdx.x;
  sh[tid] = v;
  __syncthreads();
  for (int s = blockDim.x >> 1; s > 0; s >>= 1) {
    if (tid < s) sh[tid] += sh[tid + s];
    __syncthreads();
  }
  float r = sh[0];
  __syncthreads();
  return r;
}

// ---------------------------------------------------------------------------
// LayerNorm of z2 (cols 512..1023 of h) in-place, also emits bf16 copy.
// One 256-thread block per token row, 2 elements/thread.
// ---------------------------------------------------------------------------
__global__ __launch_bounds__(256) void ln_z2_kernel(float* __restrict__ h,
    const float* __restrict__ g, const float* __restrict__ b,
    bf16_t* __restrict__ z2b) {
  __shared__ float sh[256];
  const int row = blockIdx.x, tid = threadIdx.x;
  float* z = h + (size_t)row * FFNDIM + DDIM;
  float v0 = z[tid], v1 = z[tid + 256];
  float mu  = block_reduce(v0 + v1, sh) * (1.0f / DDIM);
  float d0 = v0 - mu, d1 = v1 - mu;
  float var = block_reduce(d0 * d0 + d1 * d1, sh) * (1.0f / DDIM);
  float rstd = rsqrtf(var + LNEPS);
  float o0 = d0 * rstd * g[tid]       + b[tid];
  float o1 = d1 * rstd * g[tid + 256] + b[tid + 256];
  z[tid] = o0; z[tid + 256] = o1;
  bf16_t* zb = z2b + (size_t)row * DDIM;
  zb[tid] = (bf16_t)o0; zb[tid + 256] = (bf16_t)o1;
}

// ---------------------------------------------------------------------------
// zero_pred = LN(ReLU(ae_b1) @ ae_w2^T + ae_b2)  -- single 512-thread block
// ---------------------------------------------------------------------------
__global__ __launch_bounds__(512) void zero_pred_kernel(
    const float* __restrict__ ae_b1, const float* __restrict__ ae_w2,
    const float* __restrict__ ae_b2, const float* __restrict__ g2,
    const float* __restrict__ b2, float* __restrict__ zp) {
  __shared__ float sh[512];
  __shared__ float t0[HDIM];
  int tid = threadIdx.x;
  if (tid < HDIM) t0[tid] = fmaxf(ae_b1[tid], 0.0f);
  __syncthreads();
  float p = ae_b2[tid];
  const float* wrow = ae_w2 + (size_t)tid * HDIM;
  for (int k = 0; k < HDIM; ++k) p += t0[k] * wrow[k];
  float mu  = block_reduce(p, sh) * (1.0f / DDIM);
  float d = p - mu;
  float var = block_reduce(d * d, sh) * (1.0f / DDIM);
  zp[tid] = d * rsqrtf(var + LNEPS) * g2[tid] + b2[tid];
}

// ---------------------------------------------------------------------------
// Per-row scores + analytic softmax weights:
//   a = LN(pred_row)·z2ln_row,  o = z2ln_row·zero_pred,  δ = a-o
//   w_d = e^δ/(e^δ+1023), w_o = 1/(e^δ+1023)
// ---------------------------------------------------------------------------
__global__ __launch_bounds__(256) void score_kernel(
    const float* __restrict__ pred, const float* __restrict__ h,
    const float* __restrict__ zp, const float* __restrict__ g2,
    const float* __restrict__ b2, float* __restrict__ scal) {
  __shared__ float sh[256];
  const int row = blockIdx.x, tid = threadIdx.x;
  const float* pr = pred + (size_t)row * DDIM;
  const float* z  = h + (size_t)row * FFNDIM + DDIM;   // z2 post-LN
  float p0 = pr[tid], p1 = pr[tid + 256];
  float z0 = z[tid],  z1v = z[tid + 256];
  float mu  = block_reduce(p0 + p1, sh) * (1.0f / DDIM);
  float d0 = p0 - mu, d1 = p1 - mu;
  float var = block_reduce(d0 * d0 + d1 * d1, sh) * (1.0f / DDIM);
  float rstd = rsqrtf(var + LNEPS);
  float dp0 = d0 * rstd * g2[tid]       + b2[tid];
  float dp1 = d1 * rstd * g2[tid + 256] + b2[tid + 256];
  float a = block_reduce(dp0 * z0 + dp1 * z1v, sh);
  float o = block_reduce(z0 * zp[tid] + z1v * zp[tid + 256], sh);
  if (tid == 0) {
    float dd = fminf(fmaxf(a - o, -60.0f), 60.0f);
    float e = __expf(dd);
    float denom = e + (float)(NSEQ - 1);
    scal[2 * row]     = e / denom;      // w_d
    scal[2 * row + 1] = 1.0f / denom;   // w_o
  }
}

// ---------------------------------------------------------------------------
// S[b,d] = sum_j z1[b,j,d]  (z1 = cols 0..511 of h)
// ---------------------------------------------------------------------------
__global__ __launch_bounds__(256) void rowsum_z1_kernel(
    const float* __restrict__ h, float* __restrict__ S) {
  const int b = blockIdx.x;
  const int d = blockIdx.y * 256 + threadIdx.x;        // d < 512
  const float* base = h + (size_t)b * NSEQ * FFNDIM + d;
  float acc = 0.0f;
  for (int j = 0; j < NSEQ; ++j) acc += base[(size_t)j * FFNDIM];
  S[b * DDIM + d] = acc;
}

// ---------------------------------------------------------------------------
// attn[row,d] = (w_d - w_o)*z1[row,d] + w_o*S[b,d]   -> bf16
// ---------------------------------------------------------------------------
__global__ void attn_kernel(const float* __restrict__ h,
                            const float* __restrict__ S,
                            const float* __restrict__ scal,
                            bf16_t* __restrict__ attnb, long total) {
  long i = (long)blockIdx.x * blockDim.x + threadIdx.x;
  if (i >= total) return;
  int row = (int)(i >> 9);       // /512
  int d   = (int)(i & 511);
  int b   = row >> 10;           // /1024
  float wd = scal[2 * row], wo = scal[2 * row + 1];
  float z1v = h[(size_t)row * FFNDIM + d];
  attnb[i] = (bf16_t)((wd - wo) * z1v + wo * S[b * DDIM + d]);
}

// ---------------------------------------------------------------------------
// Host orchestration
// ---------------------------------------------------------------------------
extern "C" void kernel_launch(void* const* d_in, const int* in_sizes, int n_in,
                              void* d_out, int out_size, void* d_ws, size_t ws_size,
                              hipStream_t stream) {
  const float* x     = (const float*)d_in[0];
  const float* U_w   = (const float*)d_in[1];
  const float* U_b   = (const float*)d_in[2];
  const float* ae_w1 = (const float*)d_in[3];
  const float* ae_b1 = (const float*)d_in[4];
  const float* ae_w2 = (const float*)d_in[5];
  const float* ae_b2 = (const float*)d_in[6];
  const float* V_w   = (const float*)d_in[7];
  const float* V_b   = (const float*)d_in[8];
  const float* g1    = (const float*)d_in[9];
  const float* b1    = (const float*)d_in[10];
  const float* g2    = (const float*)d_in[11];
  const float* b2    = (const float*)d_in[12];
  float* out = (float*)d_out;

  // bump allocator on workspace
  char* ws = (char*)d_ws;
  size_t off = 0;
  auto alloc = [&](size_t bytes) -> void* {
    void* p = ws + off;
    off += (bytes + 255) & ~(size_t)255;
    return p;
  };
  bf16_t* xb    = (bf16_t*)alloc((size_t)MTOK * FDIM * 2);
  bf16_t* Uwb   = (bf16_t*)alloc((size_t)FFNDIM * FDIM * 2);
  bf16_t* w1b   = (bf16_t*)alloc((size_t)HDIM * DDIM * 2);
  bf16_t* w2b   = (bf16_t*)alloc((size_t)DDIM * HDIM * 2);
  bf16_t* Vwb   = (bf16_t*)alloc((size_t)FDIM * DDIM * 2);
  float*  hbuf  = (float*) alloc((size_t)MTOK * FFNDIM * 4);  // [z1 | z2(ln)]
  bf16_t* z2b   = (bf16_t*)alloc((size_t)MTOK * DDIM * 2);
  bf16_t* tb    = (bf16_t*)alloc((size_t)MTOK * HDIM * 2);
  float*  pred  = (float*) alloc((size_t)MTOK * DDIM * 4);
  bf16_t* attnb = (bf16_t*)alloc((size_t)MTOK * DDIM * 2);
  float*  S     = (float*) alloc((size_t)BATCH * DDIM * 4);
  float*  zp    = (float*) alloc((size_t)DDIM * 4);
  float*  scal  = (float*) alloc((size_t)MTOK * 2 * 4);

  auto cvt = [&](const float* in, bf16_t* o, long n) {
    cvt_f32_to_bf16<<<(int)((n + 255) / 256), 256, 0, stream>>>(in, o, n);
  };
  cvt(x,     xb,  (long)MTOK * FDIM);
  cvt(U_w,   Uwb, (long)FFNDIM * FDIM);
  cvt(ae_w1, w1b, (long)HDIM * DDIM);
  cvt(ae_w2, w2b, (long)DDIM * HDIM);
  cvt(V_w,   Vwb, (long)FDIM * DDIM);

  auto gemm_blocks = [](int M, int Nout) {
    return ((M >> 5) * (Nout >> 5) + 7) / 8;   // 8 waves (32x32 tiles) / block
  };

  // 1) h = GELU(x @ U_w^T + U_b)          [32768 x 1024]
  gemm_wmma_bf16<1><<<gemm_blocks(MTOK, FFNDIM), 256, 0, stream>>>(
      xb, Uwb, U_b, hbuf, nullptr, MTOK, FDIM, FFNDIM, FFNDIM);
  // 2) z2 <- LN(z2) in-place, bf16 copy
  ln_z2_kernel<<<MTOK, 256, 0, stream>>>(hbuf, g1, b1, z2b);
  // 3) t = ReLU(z2 @ ae_w1^T + ae_b1) -> bf16   [32768 x 256]
  gemm_wmma_bf16<2><<<gemm_blocks(MTOK, HDIM), 256, 0, stream>>>(
      z2b, w1b, ae_b1, nullptr, tb, MTOK, DDIM, HDIM, HDIM);
  // 4) pred = t @ ae_w2^T + ae_b2               [32768 x 512]
  gemm_wmma_bf16<0><<<gemm_blocks(MTOK, DDIM), 256, 0, stream>>>(
      tb, w2b, ae_b2, pred, nullptr, MTOK, HDIM, DDIM, DDIM);
  // 5) zero_pred (tiny)
  zero_pred_kernel<<<1, 512, 0, stream>>>(ae_b1, ae_w2, ae_b2, g2, b2, zp);
  // 6) per-row softmax weights (analytic, no NxN tensor)
  score_kernel<<<MTOK, 256, 0, stream>>>(pred, hbuf, zp, g2, b2, scal);
  // 7) S[b] = sum_j z1[b,j]
  rowsum_z1_kernel<<<dim3(BATCH, 2), 256, 0, stream>>>(hbuf, S);
  // 8) attn = (wd-wo)*z1 + wo*S -> bf16
  long total = (long)MTOK * DDIM;
  attn_kernel<<<(int)((total + 255) / 256), 256, 0, stream>>>(hbuf, S, scal, attnb, total);
  // 9) out = attn @ V_w^T + V_b                 [32768 x 256] fp32
  gemm_wmma_bf16<0><<<gemm_blocks(MTOK, FDIM), 256, 0, stream>>>(
      attnb, Vwb, V_b, out, nullptr, MTOK, DDIM, FDIM, FDIM);
}
